// SearchRankModule_34986803593849
// MI455X (gfx1250) — matrix operand order repaired and checked by reading, not compiled, not run on hardware
//
#include <hip/hip_runtime.h>

typedef __attribute__((ext_vector_type(16))) _Float16 v16h;
typedef __attribute__((ext_vector_type(8)))  _Float16 v8h;
typedef __attribute__((ext_vector_type(8)))  float    v8f;

#define NB   8
#define NC   128
#define NH   112
#define NW   112
#define NHW  (112 * 112)
#define NS   16
#define TILE 64            // pixels per block in the fused kernel
#define LDSP 136           // padded row stride (halfs) for ranked tile: 272B -> breaks bank conflicts

// ---------------------------------------------------------------------------
// Stage A: global average pool  g[b,c] = mean_{h,w} x[b,c,h,w]
// ---------------------------------------------------------------------------
__global__ __launch_bounds__(256) void pool_kernel(const float* __restrict__ x,
                                                   float* __restrict__ g) {
    __shared__ float red[256];
    const int bc = blockIdx.x;                       // 0..B*C-1
    const float* xp = x + (size_t)bc * NHW;
    float s = 0.f;
    for (int i = threadIdx.x; i < NHW; i += 256) s += xp[i];
    red[threadIdx.x] = s;
    __syncthreads();
    for (int off = 128; off > 0; off >>= 1) {
        if ((int)threadIdx.x < off) red[threadIdx.x] += red[threadIdx.x + off];
        __syncthreads();
    }
    if (threadIdx.x == 0) g[bc] = red[0] * (1.0f / (float)NHW);
}

// ---------------------------------------------------------------------------
// Stage B: ranking MLP + softmax, fold branch kernels/biases with the gate,
//          and convert fuse_w to f16 (row-major, WMMA-A friendly).
//   block 0      : MLP + softmax + fold
//   blocks 1..64 : fuse_w f32 -> f16 (256 elems each)
// ---------------------------------------------------------------------------
__global__ __launch_bounds__(256) void rank_fold_kernel(
    const float* __restrict__ g,
    const float* __restrict__ rw1, const float* __restrict__ rb1,
    const float* __restrict__ rw2, const float* __restrict__ rb2,
    const float* __restrict__ branch_w, const float* __restrict__ branch_b,
    const float* __restrict__ fuse_w,
    float* __restrict__ keff, float* __restrict__ beff,
    _Float16* __restrict__ fwh) {

    if (blockIdx.x > 0) {
        const int i = (blockIdx.x - 1) * 256 + threadIdx.x;   // 0..16383
        fwh[i] = (_Float16)fuse_w[i];
        return;
    }

    __shared__ float gs[NB * NC];
    __shared__ float hs[NB * 32];
    __shared__ float wl[NB * NS];
    const int t = threadIdx.x;

    for (int i = t; i < NB * NC; i += 256) gs[i] = g[i];
    __syncthreads();

    // h = relu(g @ rw1^T + rb1): 8*32 = 256 dot products, one per thread
    {
        const int b = t >> 5, j = t & 31;
        float acc = rb1[j];
        for (int c = 0; c < NC; ++c) acc += gs[b * NC + c] * rw1[j * NC + c];
        hs[b * 32 + j] = acc > 0.f ? acc : 0.f;
    }
    __syncthreads();

    // logits = h @ rw2^T + rb2: 8*16 = 128 dot products
    if (t < NB * NS) {
        const int b = t >> 4, s = t & 15;
        float acc = rb2[s];
        for (int j = 0; j < 32; ++j) acc += hs[b * 32 + j] * rw2[s * 32 + j];
        wl[b * NS + s] = acc;
    }
    __syncthreads();

    // softmax over S, one thread per batch
    if (t < NB) {
        float m = wl[t * NS];
        for (int s = 1; s < NS; ++s) m = fmaxf(m, wl[t * NS + s]);
        float sum = 0.f;
        float e[NS];
        for (int s = 0; s < NS; ++s) { e[s] = __expf(wl[t * NS + s] - m); sum += e[s]; }
        const float inv = 1.0f / sum;
        for (int s = 0; s < NS; ++s) wl[t * NS + s] = e[s] * inv;
    }
    __syncthreads();

    // fold: keff[b,c,k] = sum_s wts[b,s] * branch_w[s,c,k]   (9216 values)
    for (int i = t; i < NB * NC * 9; i += 256) {
        const int b = i / (NC * 9);
        const int rem = i % (NC * 9);
        const int c = rem / 9, k = rem % 9;
        float a = 0.f;
        for (int s = 0; s < NS; ++s)
            a += wl[b * NS + s] * branch_w[s * NC * 9 + c * 9 + k];
        keff[i] = a;
    }
    // fold: beff[b,c] = sum_s wts[b,s] * branch_b[s,c]
    for (int i = t; i < NB * NC; i += 256) {
        const int b = i >> 7, c = i & 127;
        float a = 0.f;
        for (int s = 0; s < NS; ++s)
            a += wl[b * NS + s] * branch_b[s * NC + c];
        beff[i] = a;
    }
}

// ---------------------------------------------------------------------------
// Stage C: per-batch-tile fused  dwconv3x3(k_eff) -> LDS(f16)  -> WMMA 1x1
//          fusion GEMM -> +fuse_b +residual -> out
// Block: 256 threads (8 waves), tile = 128 channels x 64 pixels.
// ---------------------------------------------------------------------------
__global__ __launch_bounds__(256) void fused_conv_kernel(
    const float* __restrict__ x,
    const float* __restrict__ keff, const float* __restrict__ beff,
    const _Float16* __restrict__ fwh, const float* __restrict__ fuse_b,
    float* __restrict__ out) {

    __shared__ __align__(16) _Float16 rk[TILE * LDSP];   // ranked tile, [p][c], padded

    const int tile = blockIdx.x;           // 0 .. B*196-1   (12544/64 == 196 exactly)
    const int b = tile / (NHW / TILE);
    const int p0 = (tile % (NHW / TILE)) * TILE;
    const int t = threadIdx.x;

    // ---- folded depthwise 3x3 ('SAME' zero pad), f32 math -> f16 tile -------
    {
        const int c  = t & 127;            // channel handled by this thread
        const int ph = (t >> 7) * 32;      // pixel sub-range base (0 or 32)
        const float* xb = x + (size_t)(b * NC + c) * NHW;
        const float* ke = keff + (b * NC + c) * 9;
        float kv[9];
#pragma unroll
        for (int i = 0; i < 9; ++i) kv[i] = ke[i];
        const float bv = beff[b * NC + c];

        for (int pi = 0; pi < 32; ++pi) {
            const int p = p0 + ph + pi;
            const int h = p / NW, w = p % NW;
            float acc = bv;
#pragma unroll
            for (int dh = -1; dh <= 1; ++dh) {
#pragma unroll
                for (int dw = -1; dw <= 1; ++dw) {
                    const int hh = h + dh, ww = w + dw;
                    if (hh >= 0 && hh < NH && ww >= 0 && ww < NW)
                        acc += kv[(dh + 1) * 3 + (dw + 1)] * xb[hh * NW + ww];
                }
            }
            rk[(ph + pi) * LDSP + c] = (_Float16)acc;
        }
    }
    __syncthreads();

    // ---- 1x1 fusion conv as WMMA GEMM: out[m,p] = fuse_w[m,:] . ranked[:,p] --
    const int wv   = t >> 5;               // wave 0..7 -> M rows [wv*16, wv*16+16)
    const int lane = t & 31;
    const int sel  = lane >> 4;            // half-wave select
    const int lq   = lane & 15;
    const int mb   = wv * 16;

    // A fragments (fuse_w rows, f16). ISA 16-bit A 16x32 layout:
    //   lanes 0-15 : M=lane,   VGPR0-3 K=0..7,  VGPR4-7 K=16..23
    //   lanes 16-31: M=lane-16,VGPR0-3 K=8..15, VGPR4-7 K=24..31
    v16h afrag[4];
#pragma unroll
    for (int k = 0; k < 4; ++k) {
        const _Float16* ap = fwh + (mb + lq) * NC + k * 32 + sel * 8;
        ((v8h*)&afrag[k])[0] = *(const v8h*)(ap);
        ((v8h*)&afrag[k])[1] = *(const v8h*)(ap + 16);
    }

    v8f accv[4];
    const v8f zero = {};
#pragma unroll
    for (int n = 0; n < 4; ++n) accv[n] = zero;

#pragma unroll
    for (int n = 0; n < 4; ++n) {
#pragma unroll
        for (int k = 0; k < 4; ++k) {
            // B fragment (ranked cols). 16-bit B 32x16 layout: lane holds column
            // N=lq, K = sel*8 .. sel*8+7 and 16+sel*8 .. 16+sel*8+7 within this
            // 32-K chunk (matches A) -> two 8-half runs in the [p][c] LDS tile.
            v16h bfrag;
            const _Float16* bp = &rk[(n * 16 + lq) * LDSP + k * 32 + sel * 8];
            ((v8h*)&bfrag)[0] = *(const v8h*)(bp);
            ((v8h*)&bfrag)[1] = *(const v8h*)(bp + 16);
            accv[n] = __builtin_amdgcn_wmma_f32_16x16x32_f16(
                false, afrag[k], false, bfrag, (short)0, accv[n], false, false);
        }
    }

    // ---- epilogue: + fuse_b + residual. C/D layout: lane=(m/8)*16+n, vgpr=m%8
#pragma unroll
    for (int n = 0; n < 4; ++n) {
        const int p = p0 + n * 16 + lq;
#pragma unroll
        for (int r = 0; r < 8; ++r) {
            const int m = mb + sel * 8 + r;
            const size_t idx = (size_t)(b * NC + m) * NHW + p;
            out[idx] = accv[n][r] + fuse_b[m] + x[idx];
        }
    }
}

// ---------------------------------------------------------------------------
extern "C" void kernel_launch(void* const* d_in, const int* in_sizes, int n_in,
                              void* d_out, int out_size, void* d_ws, size_t ws_size,
                              hipStream_t stream) {
    const float* x        = (const float*)d_in[0];
    const float* branch_w = (const float*)d_in[1];
    const float* branch_b = (const float*)d_in[2];
    const float* rank_w1  = (const float*)d_in[3];
    const float* rank_b1  = (const float*)d_in[4];
    const float* rank_w2  = (const float*)d_in[5];
    const float* rank_b2  = (const float*)d_in[6];
    const float* fuse_w   = (const float*)d_in[7];
    const float* fuse_b   = (const float*)d_in[8];
    float* out = (float*)d_out;

    float* ws   = (float*)d_ws;
    float* g    = ws;                 // 1024 f32
    float* keff = ws + 1152;          // 9216 f32 (leave 128 f32 gap)
    float* beff = ws + 10368;         // 1024 f32
    _Float16* fwh = (_Float16*)(ws + 11392);   // 16384 f16 (16B-aligned offset)

    pool_kernel<<<NB * NC, 256, 0, stream>>>(x, g);
    rank_fold_kernel<<<1 + (NC * NC) / 256, 256, 0, stream>>>(
        g, rank_w1, rank_b1, rank_w2, rank_b2, branch_w, branch_b, fuse_w,
        keff, beff, fwh);
    fused_conv_kernel<<<NB * (NHW / TILE), 256, 0, stream>>>(
        x, keff, beff, fwh, fuse_b, out);
}
